// ReciprocalLayerwithCNN_81398220194632
// MI455X (gfx1250) — compile-verified, matrix operand order checked by reading, not run on hardware
//
#include <hip/hip_runtime.h>
#include <hip/hip_bf16.h>

// ---------------------------------------------------------------------------
// CDNA5 (gfx1250, wave32). All matmuls via v_wmma_f32_16x16x32_bf16.
// Operands pre-converted once to bf16 in K-contiguous layouts; each wave
// computes a 32x32 tile (2x2 WMMA accumulators) -> 2 b128 loads per WMMA.
// ---------------------------------------------------------------------------

typedef __attribute__((ext_vector_type(16))) __bf16 v16bf;
typedef __attribute__((ext_vector_type(8)))  __bf16 v8bf;
typedef __attribute__((ext_vector_type(8)))  float  v8f;

__device__ __forceinline__ __bf16 f2bf(float f) {
  unsigned int u = __float_as_uint(f);
  unsigned int r = (u + 0x7FFFu + ((u >> 16) & 1u)) >> 16;  // RNE
  unsigned short s = (unsigned short)r;
  __bf16 o; __builtin_memcpy(&o, &s, 2); return o;
}

// 16-bit WMMA operand layout (ISA 7.12.2, wave32):
//   lanes 0-15 : K=k0+0..7 (v0..v3), K=k0+16..23 (v4..v7); lanes 16-31: +8.
__device__ __forceinline__ v16bf frag_bf(const __bf16* __restrict__ p, int k0) {
  int khalf = ((threadIdx.x >> 4) & 1) * 8;
  const __bf16* p0 = p + k0 + khalf;
  v8bf lo = *(const v8bf*)(p0);
  v8bf hi = *(const v8bf*)(p0 + 16);
  v16bf f;
#pragma unroll
  for (int i = 0; i < 8; ++i) { f[i] = lo[i]; f[i + 8] = hi[i]; }
  return f;
}

__device__ __forceinline__ v16bf frag_bf_guard(const __bf16* __restrict__ p, int k0, bool valid) {
  int khalf = ((threadIdx.x >> 4) & 1) * 8;
  const __bf16* p0 = p + k0 + khalf;
  v8bf lo = {}, hi = {};
  if (valid) { lo = *(const v8bf*)(p0); hi = *(const v8bf*)(p0 + 16); }
  v16bf f;
#pragma unroll
  for (int i = 0; i < 8; ++i) { f[i] = lo[i]; f[i + 8] = hi[i]; }
  return f;
}

__device__ __forceinline__ v8f wmma_bf16(v16bf a, v16bf b, v8f c) {
  return __builtin_amdgcn_wmma_f32_16x16x32_bf16(false, a, false, b, (short)0, c, false, false);
}

// Store one 16x16 f32 C/D tile with fused bias/res/relu -> f32 and/or bf16.
__device__ __forceinline__ void store_gemm(const v8f& c, int m0, int n0, int N,
                                           const float* __restrict__ bias,
                                           const float* __restrict__ res,
                                           float* __restrict__ Yf, __bf16* __restrict__ Yb,
                                           int relu) {
  int lane = threadIdx.x & 31;
  int col = n0 + (lane & 15);
  int r0  = m0 + ((lane >> 4) << 3);
  float bv = bias ? bias[col] : 0.f;
#pragma unroll
  for (int r = 0; r < 8; ++r) {
    long idx = (long)(r0 + r) * N + col;
    float v = c[r] + bv;
    if (res)  v += res[idx];
    if (relu) v = fmaxf(v, 0.f);
    if (Yf) Yf[idx] = v;
    if (Yb) Yb[idx] = f2bf(v);
  }
}

// ---------------------------------------------------------------------------
// Y = act(A[M,K]bf @ W[N,K]bf^T + bias [+ res]). Wave: 32x32 tile (2x2 WMMA).
// Block = 4 waves covering 64x64. grid = (M/64, N/64).
__global__ void gemm_wmma(const __bf16* __restrict__ A, const __bf16* __restrict__ W,
                          const float* __restrict__ bias, const float* __restrict__ res,
                          float* __restrict__ Yf, __bf16* __restrict__ Yb,
                          int N, int K, int relu) {
  int lane = threadIdx.x & 31, w = threadIdx.x >> 5;
  int m0 = blockIdx.x * 64 + (w >> 1) * 32;
  int n0 = blockIdx.y * 64 + (w & 1) * 32;
  const __bf16* ap0 = A + (long)(m0 +      (lane & 15)) * K;
  const __bf16* ap1 = A + (long)(m0 + 16 + (lane & 15)) * K;
  const __bf16* wp0 = W + (long)(n0 +      (lane & 15)) * K;
  const __bf16* wp1 = W + (long)(n0 + 16 + (lane & 15)) * K;
  v8f c00 = {}, c01 = {}, c10 = {}, c11 = {};
#pragma unroll 2
  for (int k0 = 0; k0 < K; k0 += 32) {
    v16bf a0 = frag_bf(ap0, k0), a1 = frag_bf(ap1, k0);
    v16bf b0 = frag_bf(wp0, k0), b1 = frag_bf(wp1, k0);
    c00 = wmma_bf16(a0, b0, c00); c01 = wmma_bf16(a0, b1, c01);
    c10 = wmma_bf16(a1, b0, c10); c11 = wmma_bf16(a1, b1, c11);
  }
  store_gemm(c00, m0,      n0,      N, bias, res, Yf, Yb, relu);
  store_gemm(c01, m0,      n0 + 16, N, bias, res, Yf, Yb, relu);
  store_gemm(c10, m0 + 16, n0,      N, bias, res, Yf, Yb, relu);
  store_gemm(c11, m0 + 16, n0 + 16, N, bias, res, Yf, Yb, relu);
}

// Implicit-GEMM conv1d, channels-last. X bf16 [B,L,Cin]; Wt bf16 transposed to
// [O,KT,Cin] (K-contiguous). relu fused. Wave: 32x32 tile. grid = (B*L/64, Cout/64).
__global__ void conv1d_wmma(const __bf16* __restrict__ X, const __bf16* __restrict__ Wt,
                            const float* __restrict__ bias,
                            __bf16* __restrict__ Yb, float* __restrict__ Yf, int accum,
                            int L, int Cin, int Cout, int KT, int dil) {
  int lane = threadIdx.x & 31, w = threadIdx.x >> 5;
  int m0 = blockIdx.x * 64 + (w >> 1) * 32;
  int n0 = blockIdx.y * 64 + (w & 1) * 32;
  int b  = m0 / L;
  int l0 = m0 - b * L;
  int lr = l0 + (lane & 15);
  const __bf16* wp0 = Wt + (long)(n0 +      (lane & 15)) * KT * Cin;
  const __bf16* wp1 = Wt + (long)(n0 + 16 + (lane & 15)) * KT * Cin;
  v8f c00 = {}, c01 = {}, c10 = {}, c11 = {};
  int hk = KT >> 1;
  for (int t = 0; t < KT; ++t) {
    int sl0 = lr + (t - hk) * dil, sl1 = sl0 + 16;
    bool v0 = (sl0 >= 0) && (sl0 < L), v1 = (sl1 >= 0) && (sl1 < L);
    const __bf16* ap0 = X + ((long)b * L + (v0 ? sl0 : 0)) * Cin;
    const __bf16* ap1 = X + ((long)b * L + (v1 ? sl1 : 0)) * Cin;
    const __bf16* w0 = wp0 + (long)t * Cin;
    const __bf16* w1 = wp1 + (long)t * Cin;
#pragma unroll 2
    for (int k0 = 0; k0 < Cin; k0 += 32) {
      v16bf a0 = frag_bf_guard(ap0, k0, v0), a1 = frag_bf_guard(ap1, k0, v1);
      v16bf b0 = frag_bf(w0, k0),            b1 = frag_bf(w1, k0);
      c00 = wmma_bf16(a0, b0, c00); c01 = wmma_bf16(a0, b1, c01);
      c10 = wmma_bf16(a1, b0, c10); c11 = wmma_bf16(a1, b1, c11);
    }
  }
  const v8f* cs[4] = {&c00, &c01, &c10, &c11};
#pragma unroll
  for (int ti = 0; ti < 4; ++ti) {
    int mm = l0 + ((ti >> 1) << 4), nn = n0 + ((ti & 1) << 4);
    int col = nn + (lane & 15);
    int r0  = mm + ((lane >> 4) << 3);
    float bv = bias[col];
#pragma unroll
    for (int r = 0; r < 8; ++r) {
      long idx = ((long)b * L + r0 + r) * Cout + col;
      float v = fmaxf((*cs[ti])[r] + bv, 0.f);
      if (Yb) Yb[idx] = f2bf(v);
      if (Yf) { if (accum) Yf[idx] += v; else Yf[idx] = v; }
    }
  }
}

// S[b,h,q,k] = scale * Q[b,q,h,:].K[b,k,h,:]  (D_K=64). Wave: 32x32 tile.
// grid = (Lq/64, Lk/64, B*H)
__global__ void attn_scores_wmma(const __bf16* __restrict__ Q, const __bf16* __restrict__ Km,
                                 float* __restrict__ S, int Lq, int Lk, float scale) {
  int lane = threadIdx.x & 31, w = threadIdx.x >> 5;
  int bh = blockIdx.z, b = bh >> 3, h = bh & 7;
  int q0 = blockIdx.x * 64 + (w >> 1) * 32;
  int n0 = blockIdx.y * 64 + (w & 1) * 32;
  const __bf16* qp0 = Q  + ((long)b * Lq + q0 +      (lane & 15)) * 512 + h * 64;
  const __bf16* qp1 = Q  + ((long)b * Lq + q0 + 16 + (lane & 15)) * 512 + h * 64;
  const __bf16* kp0 = Km + ((long)b * Lk + n0 +      (lane & 15)) * 512 + h * 64;
  const __bf16* kp1 = Km + ((long)b * Lk + n0 + 16 + (lane & 15)) * 512 + h * 64;
  v8f c00 = {}, c01 = {}, c10 = {}, c11 = {};
#pragma unroll
  for (int k0 = 0; k0 < 64; k0 += 32) {
    v16bf a0 = frag_bf(qp0, k0), a1 = frag_bf(qp1, k0);
    v16bf b0 = frag_bf(kp0, k0), b1 = frag_bf(kp1, k0);
    c00 = wmma_bf16(a0, b0, c00); c01 = wmma_bf16(a0, b1, c01);
    c10 = wmma_bf16(a1, b0, c10); c11 = wmma_bf16(a1, b1, c11);
  }
  long base = (long)bh * Lq * Lk;
  const v8f* cs[4] = {&c00, &c01, &c10, &c11};
#pragma unroll
  for (int ti = 0; ti < 4; ++ti) {
    int col = n0 + ((ti & 1) << 4) + (lane & 15);
    int r0  = q0 + ((ti >> 1) << 4) + ((lane >> 4) << 3);
#pragma unroll
    for (int r = 0; r < 8; ++r) S[base + (long)(r0 + r) * Lk + col] = (*cs[ti])[r] * scale;
  }
}

// O[b,q,h*64+n] = sum_k P[b,h,q,k]*V[b,k,h*64+n]. P bf16 [B,H,Lq,Lk];
// Vt bf16 [B,H,64,Lk]. Wave: 32x32 (2 waves span head dim). grid = (Lq/64,1,B*H).
__global__ void attn_av_wmma(const __bf16* __restrict__ P, const __bf16* __restrict__ Vt,
                             __bf16* __restrict__ O, int Lq, int Lk) {
  int lane = threadIdx.x & 31, w = threadIdx.x >> 5;
  int bh = blockIdx.z, b = bh >> 3, h = bh & 7;
  int q0 = blockIdx.x * 64 + (w >> 1) * 32;
  int n0 = (w & 1) * 32;
  const __bf16* ap0 = P  + ((long)bh * Lq + q0 +      (lane & 15)) * Lk;
  const __bf16* ap1 = P  + ((long)bh * Lq + q0 + 16 + (lane & 15)) * Lk;
  const __bf16* vp0 = Vt + ((long)bh * 64 + n0 +      (lane & 15)) * Lk;
  const __bf16* vp1 = Vt + ((long)bh * 64 + n0 + 16 + (lane & 15)) * Lk;
  v8f c00 = {}, c01 = {}, c10 = {}, c11 = {};
#pragma unroll 2
  for (int k0 = 0; k0 < Lk; k0 += 32) {
    v16bf a0 = frag_bf(ap0, k0), a1 = frag_bf(ap1, k0);
    v16bf b0 = frag_bf(vp0, k0), b1 = frag_bf(vp1, k0);
    c00 = wmma_bf16(a0, b0, c00); c01 = wmma_bf16(a0, b1, c01);
    c10 = wmma_bf16(a1, b0, c10); c11 = wmma_bf16(a1, b1, c11);
  }
  const v8f* cs[4] = {&c00, &c01, &c10, &c11};
#pragma unroll
  for (int ti = 0; ti < 4; ++ti) {
    int col = h * 64 + n0 + ((ti & 1) << 4) + (lane & 15);
    int r0  = q0 + ((ti >> 1) << 4) + ((lane >> 4) << 3);
#pragma unroll
    for (int r = 0; r < 8; ++r)
      O[((long)b * Lq + r0 + r) * 512 + col] = f2bf((*cs[ti])[r]);
  }
}

// Softmax, one logical row per block; strided reads supported. f32 dst + bf16 copy.
__global__ void softmax_rows(const float* __restrict__ src, float* __restrict__ dst,
                             __bf16* __restrict__ dbf, int width, long elemStride,
                             int inner, long innerStride, long outerStride,
                             long dstRowStride) {
  int row = blockIdx.x, tid = threadIdx.x;
  const float* s = src + (long)(row / inner) * outerStride + (long)(row % inner) * innerStride;
  float* d = dst + (long)row * dstRowStride;
  __bf16* db = dbf ? dbf + (long)row * width : nullptr;
  __shared__ float red[256];
  float vals[8];
  int n = (width + 255) >> 8;
  float mx = -3.4e38f;
  for (int i = 0; i < n; ++i) {
    int idx = tid + (i << 8);
    float v = (idx < width) ? s[(long)idx * elemStride] : -3.4e38f;
    vals[i] = v; mx = fmaxf(mx, v);
  }
  red[tid] = mx; __syncthreads();
  for (int o = 128; o > 0; o >>= 1) { if (tid < o) red[tid] = fmaxf(red[tid], red[tid + o]); __syncthreads(); }
  mx = red[0]; __syncthreads();
  float sum = 0.f;
  for (int i = 0; i < n; ++i) {
    int idx = tid + (i << 8);
    float e = (idx < width) ? __expf(vals[i] - mx) : 0.f;
    vals[i] = e; sum += e;
  }
  red[tid] = sum; __syncthreads();
  for (int o = 128; o > 0; o >>= 1) { if (tid < o) red[tid] += red[tid + o]; __syncthreads(); }
  float inv = 1.f / red[0];
  for (int i = 0; i < n; ++i) {
    int idx = tid + (i << 8);
    if (idx < width) {
      float v = vals[i] * inv;
      d[idx] = v;
      if (db) db[idx] = f2bf(v);
    }
  }
}

// LayerNorm width 512; writes f32 and/or bf16.
__global__ void layernorm512(const float* __restrict__ X, const float* __restrict__ g,
                             const float* __restrict__ bt, float* __restrict__ Yf,
                             __bf16* __restrict__ Yb) {
  long row = blockIdx.x; int tid = threadIdx.x;
  const float* x = X + row * 512;
  __shared__ float rs[256], rq[256];
  float a = x[tid], b = x[tid + 256];
  rs[tid] = a + b; rq[tid] = a * a + b * b;
  __syncthreads();
  for (int o = 128; o > 0; o >>= 1) { if (tid < o) { rs[tid] += rs[tid + o]; rq[tid] += rq[tid + o]; } __syncthreads(); }
  float mean = rs[0] * (1.f / 512.f);
  float var  = rq[0] * (1.f / 512.f) - mean * mean;
  float rstd = rsqrtf(var + 1e-5f);
  float y0 = (a - mean) * rstd * g[tid] + bt[tid];
  float y1 = (b - mean) * rstd * g[tid + 256] + bt[tid + 256];
  if (Yf) { Yf[row * 512 + tid] = y0; Yf[row * 512 + tid + 256] = y1; }
  if (Yb) { Yb[row * 512 + tid] = f2bf(y0); Yb[row * 512 + tid + 256] = f2bf(y1); }
}

// f32 -> bf16 (4 elems/thread). n must be a multiple of 1024.
__global__ void cvt_f32_bf16(const float* __restrict__ x, __bf16* __restrict__ y, long n) {
  long base = (long)blockIdx.x * 1024 + threadIdx.x;
#pragma unroll
  for (int i = 0; i < 4; ++i) y[base + (i << 8)] = f2bf(x[base + (i << 8)]);
}

// Conv weight [O,Cin,KT] f32 -> bf16 transposed [O,KT,Cin] (K-contiguous).
__global__ void cvt_conv_w(const float* __restrict__ w, __bf16* __restrict__ wt,
                           int Cin, int KT, long n) {
  long i = (long)blockIdx.x * 256 + threadIdx.x;
  if (i >= n) return;
  long o   = i / ((long)Cin * KT);
  long rem = i - o * (long)Cin * KT;
  int  ii  = (int)(rem / KT);
  int  t   = (int)(rem - (long)ii * KT);
  wt[(o * KT + t) * Cin + ii] = f2bf(w[i]);
}

// V bf16 [B*Lk, 512] -> Vt bf16 [B,H,64,Lk]; coalesced writes.
__global__ void transpose_v(const __bf16* __restrict__ v, __bf16* __restrict__ vt, int Lk) {
  long i = (long)blockIdx.x * 256 + threadIdx.x;   // over B*H*64*Lk
  int  k    = (int)(i % Lk);
  long rest = i / Lk;                              // b*512 + h*64 + n
  int  c    = (int)(rest & 511);
  long b    = rest >> 9;
  vt[i] = v[(b * Lk + k) * 512 + c];
}

// ---------------------------------------------------------------------------
extern "C" void kernel_launch(void* const* d_in, const int* in_sizes, int n_in,
                              void* d_out, int out_size, void* d_ws, size_t ws_size,
                              hipStream_t stream) {
  (void)in_sizes; (void)n_in; (void)out_size;
  const int B = 4, LS = 512, LP = 1024, D = 512, H = 8, DINNER = 2048;
  const int MP = B * LP, MS = B * LS;

  const float* seq_in  = (const float*)d_in[0];
  const float* prot_in = (const float*)d_in[1];
  auto P = [&](int i) { return (const float*)d_in[i]; };

  // ---- workspace layout (pointer math only; guard before any launch) ----
  char* base = (char*)d_ws; size_t off = 0;
  auto afl = [&](long n) { float*  p = (float*)(base + off);  off += (size_t)n * 4; return p; };
  auto abf = [&](long n) { __bf16* p = (__bf16*)(base + off); off += (size_t)n * 2; off = (off + 15) & ~(size_t)15; return p; };

  float*  tmp     = afl((long)MP * D);
  float*  prot_x  = afl((long)MP * D);
  float*  prot_e1 = afl((long)MP * D);
  float*  seq_e1  = afl((long)MS * D);
  float*  prot_e2 = afl((long)MP * D);
  float*  seq_e2  = afl((long)MS * D);
  float*  scores  = afl((long)B * H * LP * LS);   // raw reciprocal scores
  __bf16* seqin_b = abf((long)MS * D);
  __bf16* protin_b= abf((long)MP * D);
  __bf16* bufA_b  = abf((long)MP * D);
  __bf16* bufB_b  = abf((long)MP * D);
  __bf16* protx_b = abf((long)MP * D);
  __bf16* Qb      = abf((long)MP * D);
  __bf16* Kb      = abf((long)MP * D);
  __bf16* Vb      = abf((long)MP * D);
  __bf16* Vt      = abf((long)MP * D);
  __bf16* V2t     = abf((long)MP * D);
  __bf16* attn_b  = abf((long)B * H * LP * LP);   // shared by all 4 attn maps
  __bf16* ctx_b   = abf((long)MP * D);
  __bf16* ctx2_b  = abf((long)MS * D);
  __bf16* e1_b    = abf((long)MP * D);
  __bf16* se1_b   = abf((long)MS * D);
  __bf16* e2_b    = abf((long)MP * D);
  __bf16* se2_b   = abf((long)MS * D);
  __bf16* hid_b   = abf((long)MP * DINNER);
  __bf16* warena  = abf(19660800);                // linear + conv weights (bf16)
  if (off > ws_size) return;

  float* out     = (float*)d_out;
  float* o_prot  = out;
  float* o_seq   = out + 2097152;
  float* o_pattn = out + 3145728;
  float* o_sattn = out + 36700160;
  float* o_ps    = out + 45088768;
  float* o_sp    = out + 61865984;

  // ---- one-time conversions ----
  auto cvt = [&](const float* x, __bf16* y, long n) {
    cvt_f32_bf16<<<(unsigned)(n / 1024), 256, 0, stream>>>(x, y, n);
  };
  size_t wpos = 0;
  auto cvtLin = [&](int idx, long n) {
    __bf16* p = warena + wpos; wpos += (size_t)n; cvt(P(idx), p, n); return p;
  };
  auto cvtConv = [&](int idx, int KT) {
    long n = (long)D * D * KT; __bf16* p = warena + wpos; wpos += (size_t)n;
    cvt_conv_w<<<(unsigned)((n + 255) / 256), 256, 0, stream>>>(P(idx), p, D, KT, n);
    return p;
  };

  __bf16* cw[3][3];
  const int kt[3] = {3, 5, 7};
  for (int bi = 0; bi < 3; ++bi)
    for (int li = 0; li < 3; ++li) cw[bi][li] = cvtConv(2 + bi * 6 + li * 2, kt[bi]);
  __bf16* w_pq = cvtLin(20, 262144); __bf16* w_pk = cvtLin(22, 262144);
  __bf16* w_pv = cvtLin(24, 262144); __bf16* w_po = cvtLin(26, 262144);
  __bf16* w_sq = cvtLin(30, 262144); __bf16* w_sk = cvtLin(32, 262144);
  __bf16* w_sv = cvtLin(34, 262144); __bf16* w_so = cvtLin(36, 262144);
  __bf16* w_rq = cvtLin(40, 262144); __bf16* w_rk = cvtLin(42, 262144);
  __bf16* w_rv = cvtLin(44, 262144); __bf16* w_ro = cvtLin(46, 262144);
  __bf16* w_v2 = cvtLin(50, 262144); __bf16* w_o2 = cvtLin(52, 262144);
  __bf16* w_p1 = cvtLin(56, 1048576); __bf16* w_p2 = cvtLin(58, 1048576);
  __bf16* w_s1 = cvtLin(62, 1048576); __bf16* w_s2 = cvtLin(64, 1048576);
  cvt(seq_in,  seqin_b,  (long)MS * D);
  cvt(prot_in, protin_b, (long)MP * D);

  auto gemm = [&](const __bf16* A, const __bf16* W, const float* bias, const float* res,
                  float* Yf, __bf16* Yb, int M, int N, int K, int relu) {
    dim3 g(M / 64, N / 64, 1);
    gemm_wmma<<<g, 128, 0, stream>>>(A, W, bias, res, Yf, Yb, N, K, relu);
  };
  auto conv = [&](const __bf16* X, const __bf16* W, const float* bias,
                  __bf16* Yb, float* Yf, int accum, int KT, int dil) {
    dim3 g(MP / 64, D / 64, 1);
    conv1d_wmma<<<g, 128, 0, stream>>>(X, W, bias, Yb, Yf, accum, LP, D, D, KT, dil);
  };

  // ---- dilated CNN (branch outputs accumulate into prot_x) ----
  for (int bi = 0; bi < 3; ++bi) {
    int pb = 2 + bi * 6;
    conv(protin_b, cw[bi][0], P(pb + 1), bufA_b, nullptr, 0, kt[bi], 1);
    conv(bufA_b,   cw[bi][1], P(pb + 3), bufB_b, nullptr, 0, kt[bi], 2);
    conv(bufB_b,   cw[bi][2], P(pb + 5), nullptr, prot_x, bi > 0, kt[bi], 3);
  }
  cvt(prot_x, protx_b, (long)MP * D);

  // ---- prot self-attention ----
  gemm(protx_b, w_pq, P(21), nullptr, nullptr, Qb, MP, D, D, 0);
  gemm(protx_b, w_pk, P(23), nullptr, nullptr, Kb, MP, D, D, 0);
  gemm(protx_b, w_pv, P(25), nullptr, nullptr, Vb, MP, D, D, 0);
  transpose_v<<<(MP * D) / 256, 256, 0, stream>>>(Vb, Vt, LP);
  { dim3 g(LP / 64, LP / 64, B * H);
    attn_scores_wmma<<<g, 128, 0, stream>>>(Qb, Kb, o_pattn, LP, LP, 0.125f); }
  softmax_rows<<<B * H * LP, 256, 0, stream>>>(o_pattn, o_pattn, attn_b, LP, 1, 1 << 30, LP, 0, LP);
  { dim3 g(LP / 64, 1, B * H);
    attn_av_wmma<<<g, 128, 0, stream>>>(attn_b, Vt, ctx_b, LP, LP); }
  gemm(ctx_b, w_po, P(27), prot_x, tmp, nullptr, MP, D, D, 0);
  layernorm512<<<MP, 256, 0, stream>>>(tmp, P(28), P(29), prot_e1, e1_b);

  // ---- seq self-attention ----
  gemm(seqin_b, w_sq, P(31), nullptr, nullptr, Qb, MS, D, D, 0);
  gemm(seqin_b, w_sk, P(33), nullptr, nullptr, Kb, MS, D, D, 0);
  gemm(seqin_b, w_sv, P(35), nullptr, nullptr, Vb, MS, D, D, 0);
  transpose_v<<<(MS * D) / 256, 256, 0, stream>>>(Vb, Vt, LS);
  { dim3 g(LS / 64, LS / 64, B * H);
    attn_scores_wmma<<<g, 128, 0, stream>>>(Qb, Kb, o_sattn, LS, LS, 0.125f); }
  softmax_rows<<<B * H * LS, 256, 0, stream>>>(o_sattn, o_sattn, attn_b, LS, 1, 1 << 30, LS, 0, LS);
  { dim3 g(LS / 64, 1, B * H);
    attn_av_wmma<<<g, 128, 0, stream>>>(attn_b, Vt, ctx_b, LS, LS); }
  gemm(ctx_b, w_so, P(37), seq_in, tmp, nullptr, MS, D, D, 0);
  layernorm512<<<MS, 256, 0, stream>>>(tmp, P(38), P(39), seq_e1, se1_b);

  // ---- reciprocal cross-attention ----
  gemm(e1_b,  w_rq, P(41), nullptr, nullptr, Qb, MP, D, D, 0);
  gemm(se1_b, w_rk, P(43), nullptr, nullptr, Kb, MS, D, D, 0);
  gemm(se1_b, w_rv, P(45), nullptr, nullptr, Vb, MS, D, D, 0);
  transpose_v<<<(MS * D) / 256, 256, 0, stream>>>(Vb, Vt, LS);
  gemm(e1_b,  w_v2, P(51), nullptr, nullptr, ctx_b, MP, D, D, 0);     // V2 staging
  transpose_v<<<(MP * D) / 256, 256, 0, stream>>>(ctx_b, V2t, LP);
  { dim3 g(LP / 64, LS / 64, B * H);
    attn_scores_wmma<<<g, 128, 0, stream>>>(Qb, Kb, scores, LP, LS, 0.125f); }
  // row softmax -> ps_attn [b,h,q,k]
  softmax_rows<<<B * H * LP, 256, 0, stream>>>(scores, o_ps, attn_b, LS, 1, 1 << 30, LS, 0, LS);
  { dim3 g(LP / 64, 1, B * H);
    attn_av_wmma<<<g, 128, 0, stream>>>(attn_b, Vt, ctx_b, LP, LS); }
  gemm(ctx_b, w_ro, P(47), prot_e1, tmp, nullptr, MP, D, D, 0);
  layernorm512<<<MP, 256, 0, stream>>>(tmp, P(48), P(49), prot_e2, e2_b);
  // column softmax (transposed read of raw scores) -> sp_attn [b,h,k,q]
  softmax_rows<<<B * H * LS, 256, 0, stream>>>(scores, o_sp, attn_b, LP, LS, LS, 1, (long)LP * LS, LP);
  { dim3 g(LS / 64, 1, B * H);
    attn_av_wmma<<<g, 128, 0, stream>>>(attn_b, V2t, ctx2_b, LS, LP); }
  gemm(ctx2_b, w_o2, P(53), seq_e1, tmp, nullptr, MS, D, D, 0);
  layernorm512<<<MS, 256, 0, stream>>>(tmp, P(54), P(55), seq_e2, se2_b);

  // ---- FFNs ----
  gemm(e2_b,  w_p1, P(57), nullptr, nullptr, hid_b, MP, DINNER, D, 1);
  gemm(hid_b, w_p2, P(59), prot_e2, tmp, nullptr, MP, D, DINNER, 0);
  layernorm512<<<MP, 256, 0, stream>>>(tmp, P(60), P(61), o_prot, nullptr);
  gemm(se2_b, w_s1, P(63), nullptr, nullptr, hid_b, MS, DINNER, D, 1);
  gemm(hid_b, w_s2, P(65), seq_e2, tmp, nullptr, MS, D, DINNER, 0);
  layernorm512<<<MS, 256, 0, stream>>>(tmp, P(66), P(67), o_seq, nullptr);
}
